// TauTwoStepSimple_50723563766115
// MI455X (gfx1250) — matrix operand order repaired and checked by reading, not compiled, not run on hardware
//
#include <hip/hip_runtime.h>

// ---------------------------------------------------------------------------
// CDNA5 (gfx1250) forward pass for the ParticleNet-style tau network.
// One workgroup (8 wave32) per event. EdgeConv MLPs run on
// v_wmma_f32_16x16x32_f16 (f16 A/B, f32 accumulate), everything else VALU.
// A-fragments are register-cached across output tiles (template-fixed K).
// ---------------------------------------------------------------------------

typedef __attribute__((ext_vector_type(16))) _Float16 v16h;
typedef __attribute__((ext_vector_type(8)))  _Float16 v8h;
typedef __attribute__((ext_vector_type(8)))  float    v8f;

#define NPT   20      // points per event
#define FIN   10      // input features per point
#define KNB   16      // k nearest neighbors
#define TS    256     // LDS tile row stride (max padded K)
#define EPSF  1e-5f

struct P132  { const float* p[132]; };
struct Prep9 { const float* w[9]; };

// conv layer tables (9 layers = 3 blocks x 3 layers)
__constant__ int c_Kin [9] = { 20, 64, 64,128,128,128,256,256,128};
__constant__ int c_Nout[9] = { 64, 64, 64,128,128,128,256,128, 20};
__constant__ int c_Kpad[9] = { 32, 64, 64,128,128,128,256,256,128};
__constant__ int c_Npad[9] = { 64, 64, 64,128,128,128,256,128, 32};
__constant__ int c_Woff[9] = { 0,2048,6144,10240,26624,43008,59392,124928,157696};
__constant__ int c_cin [3] = { 10, 64, 128 };

// ---------------------------------------------------------------------------
// Prep: transpose + pad + fp16-convert all conv MLP weights into d_ws.
// Layout per layer: wT[n*Kpad + k] = w[k][n] (zero padded).
// ---------------------------------------------------------------------------
__global__ __launch_bounds__(256)
void prep_weights_kernel(Prep9 pa, _Float16* __restrict__ wT) {
  const int l   = blockIdx.y;
  const int idx = blockIdx.x * 256 + threadIdx.x;
  const int Kp = c_Kpad[l], Np = c_Npad[l];
  if (idx >= Kp * Np) return;
  const int n = idx / Kp;
  const int k = idx - n * Kp;
  float v = 0.f;
  if (k < c_Kin[l] && n < c_Nout[l]) v = pa.w[l][k * c_Nout[l] + n];
  wT[c_Woff[l] + idx] = (_Float16)v;
}

// ---------------------------------------------------------------------------
// Small device helpers
// ---------------------------------------------------------------------------
__device__ __forceinline__ float eluf(float x) {
  return x > 0.f ? x : (expf(x) - 1.f);
}

// One MLP layer of an EdgeConv block, compile-time padded dims.
// cur: 16 x KP fp16 tile (row stride TS) in LDS. nxt: 16 x NP output tile.
// Each wave caches its A-fragments in registers (KP/32 x v16h) and iterates
// over its share of the NP/16 output column tiles.
template <int KP, int NP>
__device__ __forceinline__ void mlp_layer(
    const _Float16* __restrict__ cur, _Float16* __restrict__ nxt,
    const _Float16* __restrict__ W, int Na,
    const float* __restrict__ bnb, const float* __restrict__ bng,
    const float* __restrict__ bnm, const float* __restrict__ bnv,
    int lane, int wave) {
  constexpr int NKT = KP / 32;   // K-tiles of 32
  constexpr int NNT = NP / 16;   // N-tiles of 16
  const int mrow = lane & 15;
  const int hif  = lane >> 4;

  // Register-cached A fragments (shared across all nt iterations).
  v16h afr[NKT];
  const _Float16* arow = cur + mrow * TS + hif * 16;
#pragma unroll
  for (int kt = 0; kt < NKT; kt++) {
    const v8h alo = *(const v8h*)(arow + kt * 32);       // K {0..7}/{16..23}
    const v8h ahi = *(const v8h*)(arow + kt * 32 + 8);   // K {8..15}/{24..31}
    afr[kt] = __builtin_shufflevector(
        alo, ahi, 0, 1, 2, 3, 4, 5, 6, 7, 8, 9, 10, 11, 12, 13, 14, 15);
  }

  for (int nt = wave; nt < NNT; nt += 8) {               // wave-uniform bounds
    v8f acc = {0.f, 0.f, 0.f, 0.f, 0.f, 0.f, 0.f, 0.f};
    const int ncol = nt * 16 + mrow;
    const _Float16* bcol = W + (size_t)ncol * KP + hif * 16;
#pragma unroll
    for (int kt = 0; kt < NKT; kt++) {
      const v16h bv = *(const v16h*)(bcol + kt * 32);    // contiguous column chunk
      acc = __builtin_amdgcn_wmma_f32_16x16x32_f16(
          false, afr[kt], false, bv, (short)0, acc, false, false);
    }
    // fused BatchNorm + ReLU epilogue, fp16 store for the next layer's A tile
    const bool valid = (ncol < Na);
    float sc = 0.f, sb = 0.f, sm = 0.f;
    if (valid) {
      sc = bng[ncol] * rsqrtf(bnv[ncol] + EPSF);
      sb = bnb[ncol];
      sm = bnm[ncol];
    }
#pragma unroll
    for (int r = 0; r < 8; r++) {
      const int m = r + hif * 8;            // C/D layout: M = r + 8*(lane>=16)
      const float y = valid ? fmaxf((acc[r] - sm) * sc + sb, 0.f) : 0.f;
      nxt[m * TS + ncol] = (_Float16)y;
    }
  }
}

// LayerNorm of x[0..L) into y, using red[512] scratch. Block-wide.
__device__ void lnorm(const float* __restrict__ x, int L,
                      const float* __restrict__ g, const float* __restrict__ b,
                      float* __restrict__ y, float* __restrict__ red) {
  const int tid = threadIdx.x;
  float s = 0.f, s2 = 0.f;
  for (int j = tid; j < L; j += 256) { float v = x[j]; s += v; s2 += v * v; }
  red[tid] = s; red[256 + tid] = s2;
  __syncthreads();
  for (int off = 128; off > 0; off >>= 1) {
    if (tid < off) {
      red[tid]       += red[tid + off];
      red[256 + tid] += red[256 + tid + off];
    }
    __syncthreads();
  }
  const float mu   = red[0] / (float)L;
  const float var  = red[256] / (float)L - mu * mu;
  const float rinv = rsqrtf(var + EPSF);
  for (int j = tid; j < L; j += 256) y[j] = (x[j] - mu) * rinv * g[j] + b[j];
  __syncthreads();
}

// FFN per reference apply_ffn: 4x [elu(ln(x)@W+b)], then elu(ln4(x)),
// then ln5(x)@W4+b4. Params: lin i -> b: pb+2i, w: pb+2i+1; ln i -> b: pb+10+2i, g: pb+11+2i.
__device__ void ffn_run(const P132& P, int pb, int din, int w, int dout,
                        const float* __restrict__ xin,
                        float* __restrict__ xa, float* __restrict__ xb,
                        float* __restrict__ red, float* __restrict__ outp) {
  const int tid = threadIdx.x;
  for (int t = tid; t < din; t += 256) xa[t] = xin[t];
  __syncthreads();
  int L = din;
  for (int i = 0; i < 4; i++) {
    lnorm(xa, L, P.p[pb + 11 + 2 * i], P.p[pb + 10 + 2 * i], xb, red);
    const float* W  = P.p[pb + 2 * i + 1];
    const float* bv = P.p[pb + 2 * i];
    for (int o = tid; o < w; o += 256) {
      float s = bv[o];
      for (int j = 0; j < L; j++) s += xb[j] * W[j * w + o];
      xa[o] = eluf(s);
    }
    __syncthreads();
    L = w;
  }
  lnorm(xa, L, P.p[pb + 19], P.p[pb + 18], xb, red);
  for (int t = tid; t < L; t += 256) xb[t] = eluf(xb[t]);
  __syncthreads();
  lnorm(xb, L, P.p[pb + 21], P.p[pb + 20], xa, red);
  const float* W  = P.p[pb + 9];
  const float* bv = P.p[pb + 8];
  for (int o = tid; o < dout; o += 256) {
    float s = bv[o];
    for (int j = 0; j < L; j++) s += xa[j] * W[j * dout + o];
    outp[o] = s;
  }
  __syncthreads();
}

// ---------------------------------------------------------------------------
// Main kernel: one block (256 threads, 8 wave32) per event.
// ---------------------------------------------------------------------------
__global__ __launch_bounds__(256)
void tau_forward_kernel(P132 P, const _Float16* __restrict__ wT,
                        float* __restrict__ out, int Bev) {
  __shared__ __align__(32) _Float16 tiles[2 * 16 * TS];  // ping/pong edge tiles (fp16)
  __shared__ float ftsA[NPT * 128];
  __shared__ float ftsB[NPT * 128];
  __shared__ float pos_s[NPT * 3];
  __shared__ float d2s[NPT * NPT];
  __shared__ int   knn_s[NPT * KNB];
  __shared__ float xa[600];
  __shared__ float xb[600];
  __shared__ float encv[64];
  __shared__ float red[512];

  const int tid  = threadIdx.x;
  const int lane = tid & 31;
  const int wave = tid >> 5;
  const int e    = blockIdx.x;

  const float* gf  = P.p[0] + (size_t)e * (NPT * FIN);
  const float* pos = P.p[1] + (size_t)e * (NPT * 3);

  // input BN -> ftsA (stride 128)
  {
    const float* ib = P.p[84]; const float* ig = P.p[85];
    const float* im = P.p[86]; const float* iv = P.p[87];
    for (int t = tid; t < NPT * FIN; t += 256) {
      const int c = t % FIN;
      ftsA[(t / FIN) * 128 + c] =
          (gf[t] - im[c]) * rsqrtf(iv[c] + EPSF) * ig[c] + ib[c];
    }
    for (int t = tid; t < NPT * 3; t += 256) pos_s[t] = pos[t];
  }
  __syncthreads();

  float* fcur = ftsA;
  float* fnxt = ftsB;

  for (int b = 0; b < 3; b++) {
    const int cin   = c_cin[b];
    const int cbase = 2 + 20 * b;

    // ---- pairwise squared distances (VALU; tiny) ----
    const float* pts = (b == 0) ? pos_s : fcur;
    const int pdim   = (b == 0) ? 3 : cin;
    const int pstr   = (b == 0) ? 3 : 128;
    for (int t = tid; t < NPT * NPT; t += 256) {
      const int i = t / NPT, j = t % NPT;
      float s = 0.f;
      for (int c = 0; c < pdim; c++) {
        const float d = pts[i * pstr + c] - pts[j * pstr + c];
        s += d * d;
      }
      d2s[t] = (i == j) ? 1e9f : s;
    }
    __syncthreads();
    // ---- stable-rank kNN selection (order irrelevant under max-aggr) ----
    for (int t = tid; t < NPT * NPT; t += 256) {
      const int i = t / NPT, j = t % NPT;
      const float dij = d2s[i * NPT + j];
      int rank = 0;
      for (int j2 = 0; j2 < NPT; j2++) {
        const float d2v = d2s[i * NPT + j2];
        rank += (d2v < dij) || (d2v == dij && j2 < j);
      }
      if (rank < KNB) knn_s[i * KNB + rank] = j;
    }
    __syncthreads();

    const int K0 = c_Kpad[b * 3 + 0];
    for (int p = 0; p < NPT; p++) {
      _Float16* cur = tiles;
      _Float16* nxt = tiles + 16 * TS;
      // ---- build edge tile h0: 16 rows (neighbors) x [xi | xj-xi | pad] ----
      for (int t = tid; t < 16 * K0; t += 256) {
        const int r = t / K0, c = t - r * K0;
        float v = 0.f;
        if (c < cin) {
          v = fcur[p * 128 + c];
        } else if (c < 2 * cin) {
          const int j = knn_s[p * KNB + r];
          v = fcur[j * 128 + (c - cin)] - fcur[p * 128 + (c - cin)];
        }
        cur[r * TS + c] = (_Float16)v;
      }
      __syncthreads();

      // ---- 3-layer MLP via WMMA, fused BN+ReLU epilogue ----
      for (int l = 0; l < 3; l++) {
        const int li = b * 3 + l;
        const int Na = c_Nout[li];
        const _Float16* W = wT + c_Woff[li];
        const float* bnb = P.p[cbase + 4 * l + 0];
        const float* bng = P.p[cbase + 4 * l + 1];
        const float* bnm = P.p[cbase + 4 * l + 2];
        const float* bnv = P.p[cbase + 4 * l + 3];
        switch (li) {
          case 0:
            mlp_layer< 32,  64>(cur, nxt, W, Na, bnb, bng, bnm, bnv, lane, wave);
            break;
          case 1: case 2:
            mlp_layer< 64,  64>(cur, nxt, W, Na, bnb, bng, bnm, bnv, lane, wave);
            break;
          case 3: case 4: case 5:
            mlp_layer<128, 128>(cur, nxt, W, Na, bnb, bng, bnm, bnv, lane, wave);
            break;
          case 6:
            mlp_layer<256, 256>(cur, nxt, W, Na, bnb, bng, bnm, bnv, lane, wave);
            break;
          case 7:
            mlp_layer<256, 128>(cur, nxt, W, Na, bnb, bng, bnm, bnv, lane, wave);
            break;
          default:
            mlp_layer<128,  32>(cur, nxt, W, Na, bnb, bng, bnm, bnv, lane, wave);
            break;
        }
        __syncthreads();
        _Float16* t0 = cur; cur = nxt; nxt = t0;
      }

      // ---- max over neighbors + BN'd skip + ReLU -> fnxt ----
      {
        const int Na2 = c_Nout[b * 3 + 2];
        const float* wsk = P.p[cbase + 19];
        const float* skb = P.p[cbase + 12];
        const float* skg = P.p[cbase + 13];
        const float* skm = P.p[cbase + 14];
        const float* skv = P.p[cbase + 15];
        for (int n = tid; n < Na2; n += 256) {
          float mx = -1e30f;
          for (int r = 0; r < KNB; r++) mx = fmaxf(mx, (float)cur[r * TS + n]);
          float s = 0.f;
          for (int c = 0; c < cin; c++) s += fcur[p * 128 + c] * wsk[c * Na2 + n];
          s = (s - skm[n]) * rsqrtf(skv[n] + EPSF) * skg[n] + skb[n];
          fnxt[p * 128 + n] = fmaxf(mx + s, 0.f);
        }
      }
      __syncthreads();
    }
    { float* t = fcur; fcur = fnxt; fnxt = t; }
    __syncthreads();
  }

  // ---- heads. fcur holds final fts (20 x 20, stride 128) ----
  for (int t = tid; t < NPT; t += 256) {
    float s = 0.f;
    for (int pp = 0; pp < NPT; pp++) s += fcur[pp * 128 + t];
    s *= (1.f / (float)NPT);
    out[(size_t)e * NPT + t] = 1.f / (1.f + expf(-s));
  }
  for (int t = tid; t < 600; t += 256) {
    const int pp = t / 30, c = t - pp * 30;
    xa[t] = (c < FIN) ? gf[pp * FIN + c] : fcur[pp * 128 + (c - FIN)];
  }
  __syncthreads();

  ffn_run(P, 62, 600, 128, 64, xa, xa, xb, red, encv);                     // enc
  ffn_run(P, 88, 64, 64, 1, encv, xa, xb, red,
          out + (size_t)Bev * NPT + e);                                    // istau
  ffn_run(P, 110, 64, 64, 4, encv, xa, xb, red,
          out + (size_t)Bev * 21 + (size_t)e * 4);                         // p4
}

// ---------------------------------------------------------------------------
// Host launcher.
// Assumed pytree (sorted-key) flattening of setup_inputs():
//   0 gnnfeats, 1 gnnpos, then params:
//   convs c: base=2+20c: bns[l]{b,g,m,v} -> base+4l.., bnskip{b,g,m,v} -> base+12..,
//            ws[l] -> base+16+l, wskip -> base+19
//   enc  : 62..83  (lin b/w x5, then ln b/g x6)
//   input_bn: 84..87 (b,g,m,v)
//   istau: 88..109, p4: 110..131
// ---------------------------------------------------------------------------
extern "C" void kernel_launch(void* const* d_in, const int* in_sizes, int n_in,
                              void* d_out, int out_size, void* d_ws, size_t ws_size,
                              hipStream_t stream) {
  (void)out_size; (void)ws_size;
  const int Bev = in_sizes[0] / (NPT * FIN);

  P132 P;
  for (int i = 0; i < 132; i++)
    P.p[i] = (const float*)((i < n_in) ? d_in[i] : d_in[0]);

  Prep9 pr;
  for (int c = 0; c < 3; c++)
    for (int l = 0; l < 3; l++)
      pr.w[c * 3 + l] = (const float*)d_in[2 + 20 * c + 16 + l];

  _Float16* wT = (_Float16*)d_ws;   // 161792 fp16 elems = ~316 KB

  // weight transpose/pad/convert (largest layer 256x256 -> 65536 elems)
  hipLaunchKernelGGL(prep_weights_kernel, dim3(256, 9), dim3(256), 0, stream,
                     pr, wT);

  hipLaunchKernelGGL(tau_forward_kernel, dim3(Bev), dim3(256), 0, stream,
                     P, (const _Float16*)wT, (float*)d_out, Bev);
}
